// server_hgcn_7997229105406
// MI455X (gfx1250) — matrix operand order repaired and checked by reading, not compiled, not run on hardware
//
#include <hip/hip_runtime.h>

#define HG_N    100000
#define HG_E    1600000
#define HG_D    64
#define HG_B    64
#define HG_K1   50000
#define HG_K2   25000
#define HG_EPS  1e-7f
#define HG_LAMB 1.0f
#define HG_MAXN (1.0f - 1e-5f)
#define ENC_NEGINF 0x007FFFFFu

typedef __attribute__((ext_vector_type(16))) _Float16 v16h;
typedef __attribute__((ext_vector_type(8)))  _Float16 v8h;
typedef __attribute__((ext_vector_type(8)))  float    v8f;
typedef __attribute__((ext_vector_type(4)))  unsigned u32x4;
typedef __attribute__((ext_vector_type(4)))  int      i32x4;
typedef __attribute__((ext_vector_type(8)))  int      i32x8;

// ---------- device helpers ----------
__device__ inline float wsum(float v) {
#pragma unroll
  for (int m = 16; m; m >>= 1) v += __shfl_xor(v, m, 32);
  return v;
}
__device__ inline float artanh_c(float x) {
  x = fminf(x, 1.0f - 1e-7f);
  x = fmaxf(x, -1.0f + 1e-7f);
  return 0.5f * logf((1.0f + x) / (1.0f - x));
}
// order-preserving float<->uint encoding for atomicMax on floats
__device__ inline unsigned fenc(float f) {
  unsigned u = __float_as_uint(f);
  return (u & 0x80000000u) ? ~u : (u | 0x80000000u);
}
__device__ inline float fdec(unsigned e) {
  return (e & 0x80000000u) ? __uint_as_float(e & 0x7FFFFFFFu)
                           : __uint_as_float(~e);
}

// ---------- fills ----------
__global__ void k_fill_f32(float* p, float v, int n) {
  int i = blockIdx.x * 256 + threadIdx.x;
  if (i < n) p[i] = v;
}
__global__ void k_fill_u32(unsigned* p, unsigned v, int n) {
  int i = blockIdx.x * 256 + threadIdx.x;
  if (i < n) p[i] = v;
}

// ---------- tiny precompute ----------
__global__ void k_w2h(const float* __restrict__ w, _Float16* __restrict__ wh) {
  int i = blockIdx.x * 256 + threadIdx.x;
  if (i < HG_D * HG_D) wh[i] = (_Float16)w[i];
}
// hb = ball_proj(expmap0(b))  (one wave)
__global__ void k_hb(const float* __restrict__ b, float* __restrict__ hb) {
  int lane = threadIdx.x & 31;
  float a0 = b[lane * 2], a1 = b[lane * 2 + 1];
  float nrm = fmaxf(sqrtf(wsum(a0 * a0 + a1 * a1)), HG_EPS);
  float s = tanhf(nrm) / nrm;
  float u0 = a0 * s, u1 = a1 * s;
  float un = tanhf(nrm);
  float pr = (un > HG_MAXN) ? HG_MAXN / un : 1.0f;
  hb[lane * 2] = u0 * pr;
  hb[lane * 2 + 1] = u1 * pr;
}

// ---------- row-wise hyperbolic maps (1 wave32 per 64-wide row) ----------
__global__ void k_exp_proj(const float* __restrict__ in, float* __restrict__ out, int n) {
  int lane = threadIdx.x & 31;
  int row = blockIdx.x * 8 + (threadIdx.x >> 5);
  if (row >= n) return;
  const float* ip = in + (size_t)row * HG_D + lane * 2;
  float a0 = ip[0], a1 = ip[1];
  float nrm = fmaxf(sqrtf(wsum(a0 * a0 + a1 * a1)), HG_EPS);
  float s = tanhf(nrm) / nrm;
  float un = tanhf(nrm);
  float pr = (un > HG_MAXN) ? HG_MAXN / un : 1.0f;
  float* op = out + (size_t)row * HG_D + lane * 2;
  op[0] = a0 * s * pr;
  op[1] = a1 * s * pr;
}
__global__ void k_logmap_both(const float* __restrict__ in, float* __restrict__ o32,
                              _Float16* __restrict__ o16, int n) {
  int lane = threadIdx.x & 31;
  int row = blockIdx.x * 8 + (threadIdx.x >> 5);
  if (row >= n) return;
  const float* ip = in + (size_t)row * HG_D + lane * 2;
  float a0 = ip[0], a1 = ip[1];
  float nrm = fmaxf(sqrtf(wsum(a0 * a0 + a1 * a1)), HG_EPS);
  float s = artanh_c(nrm) / nrm;
  float t0 = a0 * s, t1 = a1 * s;
  float* op = o32 + (size_t)row * HG_D + lane * 2;
  _Float16* hp = o16 + (size_t)row * HG_D + lane * 2;
  op[0] = t0; op[1] = t1;
  hp[0] = (_Float16)t0; hp[1] = (_Float16)t1;
}
// HypLinear tail: mv=proj(exp(gemm)); h=proj(mobius_add(mv,hb)); ht16=f16(logmap0(h))
__global__ void k_hl_post(const float* __restrict__ mv, const float* __restrict__ hb,
                          _Float16* __restrict__ ht16, int n) {
  int lane = threadIdx.x & 31;
  int row = blockIdx.x * 8 + (threadIdx.x >> 5);
  if (row >= n) return;
  const float* ip = mv + (size_t)row * HG_D + lane * 2;
  float a0 = ip[0], a1 = ip[1];
  float nrm = fmaxf(sqrtf(wsum(a0 * a0 + a1 * a1)), HG_EPS);
  float s = tanhf(nrm) / nrm;
  float un = tanhf(nrm);
  float pr = (un > HG_MAXN) ? HG_MAXN / un : 1.0f;
  float x0 = a0 * s * pr, x1 = a1 * s * pr;
  float y0 = hb[lane * 2], y1 = hb[lane * 2 + 1];
  float x2 = wsum(x0 * x0 + x1 * x1);
  float y2 = wsum(y0 * y0 + y1 * y1);
  float xy = wsum(x0 * y0 + x1 * y1);
  float A = 1.0f + 2.0f * xy + y2;
  float Bc = 1.0f - x2;
  float den = fmaxf(1.0f + 2.0f * xy + x2 * y2, HG_EPS);
  float h0 = (A * x0 + Bc * y0) / den, h1 = (A * x1 + Bc * y1) / den;
  float nh = sqrtf(wsum(h0 * h0 + h1 * h1));
  float pr2 = (nh > HG_MAXN) ? HG_MAXN / nh : 1.0f;
  h0 *= pr2; h1 *= pr2;
  float nh2 = fmaxf(nh * pr2, HG_EPS);
  float ls = artanh_c(nh2) / nh2;
  _Float16* op = ht16 + (size_t)row * HG_D + lane * 2;
  op[0] = (_Float16)(h0 * ls);
  op[1] = (_Float16)(h1 * ls);
}
// HypAgg tail + HypAct
__global__ void k_agg_act(const float* __restrict__ agg, float* __restrict__ out, int n) {
  int lane = threadIdx.x & 31;
  int row = blockIdx.x * 8 + (threadIdx.x >> 5);
  if (row >= n) return;
  const float* ip = agg + (size_t)row * HG_D + lane * 2;
  float a0 = ip[0], a1 = ip[1];
  float nrm = fmaxf(sqrtf(wsum(a0 * a0 + a1 * a1)), HG_EPS);
  float s = tanhf(nrm) / nrm;
  float un = tanhf(nrm);
  float pr = (un > HG_MAXN) ? HG_MAXN / un : 1.0f;
  float h0 = a0 * s * pr, h1 = a1 * s * pr;
  float nh = fmaxf(sqrtf(wsum(h0 * h0 + h1 * h1)), HG_EPS);
  float ls = artanh_c(nh) / nh;
  float r0 = fmaxf(h0 * ls, 0.0f), r1 = fmaxf(h1 * ls, 0.0f);
  float nr = fmaxf(sqrtf(wsum(r0 * r0 + r1 * r1)), HG_EPS);
  float s2 = tanhf(nr) / nr;
  float un2 = tanhf(nr);
  float pr2 = (un2 > HG_MAXN) ? HG_MAXN / un2 : 1.0f;
  float* op = out + (size_t)row * HG_D + lane * 2;
  op[0] = r0 * s2 * pr2;
  op[1] = r1 * s2 * pr2;
}

// ---------- WMMA GEMM: out[n,64] = T16[n,64] @ W^T (W row-major [64,64] f16) ----------
// nrows is always a multiple of 16 in this pipeline.
__global__ void k_gemm_wmma(const _Float16* __restrict__ T, const _Float16* __restrict__ Wh,
                            float* __restrict__ out, int nrows) {
  __shared__ _Float16 sW[HG_D * HG_D];

#if defined(__gfx1250__) && __has_builtin(__builtin_amdgcn_tensor_load_to_lds)
  // Stage the 8KB weight tile with the Tensor Data Mover: one DMA per block,
  // issued by wave 0 only (TDM ignores EXEC, so gate with a uniform branch).
  if ((threadIdx.x >> 5) == 0) {
    unsigned long long ga = (unsigned long long)(const void*)Wh;
    u32x4 g0;
    g0.x = 1u;                                   // count=1, user descriptor
    g0.y = 0u;                                   // lds_addr: sW at LDS offset 0
    g0.z = (unsigned)(ga & 0xFFFFFFFFull);       // global_addr[31:0]
    g0.w = (unsigned)((ga >> 32) & 0x01FFFFFFull) | (2u << 30);  // addr[56:32] | type=2
    i32x8 g1;
    g1[0] = (int)(1u << 16);                     // data_size=1 (2 bytes/elem)
    g1[1] = (int)(64u << 16);                    // tensor_dim0 = 64 (lo16 @ [31:16])
    g1[2] = (int)(64u << 16);                    // dim0 hi16=0 | tensor_dim1=64 lo16
    g1[3] = (int)(64u << 16);                    // dim1 hi16=0 | tile_dim0 = 64
    g1[4] = 64;                                  // tile_dim1 = 64, tile_dim2 = 0
    g1[5] = 64;                                  // tensor_dim0_stride = 64 (lo32)
    g1[6] = 0;                                   // stride hi | dim1_stride lo
    g1[7] = 0;
    i32x4 gz = {0, 0, 0, 0};                     // 2D tensor: groups 2/3 unused
#if __clang_major__ >= 23
    i32x8 gz8 = {0, 0, 0, 0, 0, 0, 0, 0};
    __builtin_amdgcn_tensor_load_to_lds(g0, g1, gz, gz, gz8, 0);
#else
    __builtin_amdgcn_tensor_load_to_lds(g0, g1, gz, gz, 0);
#endif
    __builtin_amdgcn_s_wait_tensorcnt(0);
  }
#else
  for (int i = threadIdx.x; i < (HG_D * HG_D) / 8; i += 256)
    *((v8h*)sW + i) = *((const v8h*)Wh + i);
#endif
  __syncthreads();

  int wave = threadIdx.x >> 5;
  int lane = threadIdx.x & 31;
  int tile = blockIdx.x * 8 + wave;         // one 16-row tile per wave
  int row0 = tile * 16;
  if (row0 >= nrows) return;
  int r = lane & 15, hi = lane >> 4;

  // A fragments per ISA 16-bit A 16x32 layout (lanes 0-15: K 0-7,16-23; lanes 16-31: K 8-15,24-31)
  v16h a[2];
  const _Float16* tp = T + (size_t)(row0 + r) * HG_D;
#pragma unroll
  for (int kt = 0; kt < 2; ++kt) {
    int k0 = kt * 32 + hi * 8;
    v8h lo = *(const v8h*)(tp + k0);
    v8h hh = *(const v8h*)(tp + k0 + 16);
#pragma unroll
    for (int e = 0; e < 8; ++e) { a[kt][e] = lo[e]; a[kt][e + 8] = hh[e]; }
  }
#pragma unroll
  for (int ct = 0; ct < 4; ++ct) {
    int col = ct * 16 + r;
    v8f c = {};
#pragma unroll
    for (int kt = 0; kt < 2; ++kt) {
      // B per 16-bit 32x16 layout: lanes 0-15 K 0-7,16-23; lanes 16-31 K 8-15,24-31, col = lane&15
      int kb = kt * 32 + hi * 8;
      v8h blo = *(const v8h*)(sW + col * HG_D + kb);
      v8h bhi = *(const v8h*)(sW + col * HG_D + kb + 16);
      v16h b;
#pragma unroll
      for (int e = 0; e < 8; ++e) { b[e] = blo[e]; b[e + 8] = bhi[e]; }
      c = __builtin_amdgcn_wmma_f32_16x16x32_f16(false, a[kt], false, b,
                                                 (short)0, c, false, false);
    }
    // C/D layout: lane<16 -> M=v, lane>=16 -> M=v+8 ; N = lane&15 (rows always in range)
#pragma unroll
    for (int v = 0; v < 8; ++v)
      out[(size_t)(row0 + v + 8 * hi) * HG_D + col] = c[v];
  }
}

// ---------- graph kernels ----------
__global__ void k_deg(const int* __restrict__ dst, const float* __restrict__ w,
                      float* __restrict__ deg, int nE) {
  int i = blockIdx.x * 256 + threadIdx.x;
  if (i < nE) atomicAdd(&deg[dst[i]], w[i]);
}
// one wave per edge; gather f16 row of src, scaled atomic add into dst row
__global__ void k_scatter(const int* __restrict__ src, const int* __restrict__ dst,
                          const float* __restrict__ w, const float* __restrict__ deg,
                          const _Float16* __restrict__ ht, float* __restrict__ agg, int nE) {
  int e = blockIdx.x * 8 + (threadIdx.x >> 5);
  if (e >= nE) return;
  int lane = threadIdx.x & 31;
  int s = src[e], d = dst[e];
  float coef = w[e] / deg[d];               // deg pre-seeded with EPS
  const _Float16* hp = ht + (size_t)s * HG_D + lane * 2;
  __builtin_prefetch((const void*)(src + e + 64), 0, 0);  // warm index stream
  float f0 = (float)hp[0], f1 = (float)hp[1];
  float* ap = agg + (size_t)d * HG_D + lane * 2;
  atomicAdd(ap, coef * f0);
  atomicAdd(ap + 1, coef * f1);
}
// score = sum|xt-agg| ; ps = xt.a_s ; pd = xt.a_d
__global__ void k_score_proj(const float* __restrict__ xt, const float* __restrict__ agg,
                             const float* __restrict__ as, const float* __restrict__ ad,
                             unsigned* __restrict__ score, float* __restrict__ ps,
                             float* __restrict__ pd, int n) {
  int lane = threadIdx.x & 31;
  int row = blockIdx.x * 8 + (threadIdx.x >> 5);
  if (row >= n) return;
  const float* tp = xt + (size_t)row * HG_D + lane * 2;
  const float* gp = agg + (size_t)row * HG_D + lane * 2;
  float t0 = tp[0], t1 = tp[1], g0 = gp[0], g1 = gp[1];
  float a0 = as[lane * 2], a1 = as[lane * 2 + 1];
  float d0 = ad[lane * 2], d1 = ad[lane * 2 + 1];
  float sc = wsum(fabsf(t0 - g0) + fabsf(t1 - g1));
  float p0 = wsum(t0 * a0 + t1 * a1);
  float p1 = wsum(t0 * d0 + t1 * d1);
  if (lane == 0) {
    score[row] = __float_as_uint(sc);   // sc >= 0 -> bits are order-preserving
    ps[row] = p0;
    pd[row] = p1;
  }
}

// ---------- radix top-k (16+16 bit) ----------
__global__ void k_hist_hi(const unsigned* __restrict__ s, unsigned* __restrict__ h, int n) {
  int i = blockIdx.x * 256 + threadIdx.x;
  if (i < n) atomicAdd(&h[s[i] >> 16], 1u);
}
__global__ void k_sel_hi(const unsigned* __restrict__ h, unsigned* sel, int k) {
  if (threadIdx.x || blockIdx.x) return;
  unsigned cum = 0;
  for (int b = 65535; b >= 0; --b) {
    unsigned c = h[b];
    if (cum + c >= (unsigned)k) { sel[0] = (unsigned)b; sel[1] = cum; return; }
    cum += c;
  }
  sel[0] = 0; sel[1] = cum;
}
__global__ void k_hist_lo(const unsigned* __restrict__ s, const unsigned* __restrict__ sel,
                          unsigned* __restrict__ h, int n) {
  int i = blockIdx.x * 256 + threadIdx.x;
  if (i >= n) return;
  unsigned v = s[i];
  if ((v >> 16) == sel[0]) atomicAdd(&h[v & 0xFFFFu], 1u);
}
__global__ void k_sel_lo(const unsigned* __restrict__ h, unsigned* sel, int k) {
  if (threadIdx.x || blockIdx.x) return;
  unsigned above = sel[1];
  for (int b = 65535; b >= 0; --b) {
    unsigned c = h[b];
    if (above + c >= (unsigned)k) {
      sel[2] = (sel[0] << 16) | (unsigned)b;   // threshold bits
      sel[3] = (unsigned)k - above;            // ties to keep (index order)
      return;
    }
    above += c;
  }
  sel[2] = 0; sel[3] = 0;
}
__global__ void k_flags(const unsigned* __restrict__ s, const unsigned* __restrict__ sel,
                        unsigned* __restrict__ gt, unsigned* __restrict__ eq, int n) {
  int i = blockIdx.x * 256 + threadIdx.x;
  if (i >= n) return;
  unsigned v = s[i], T = sel[2];
  gt[i] = (v > T) ? 1u : 0u;
  eq[i] = (v == T) ? 1u : 0u;
}

// exclusive prefix-scan: per-block, block-sums, add
__global__ void k_scan1(const unsigned* __restrict__ in, unsigned* __restrict__ out,
                        unsigned* __restrict__ bsum, int n) {
  __shared__ unsigned sh[256];
  int i = blockIdx.x * 256 + threadIdx.x;
  unsigned v = (i < n) ? in[i] : 0u;
  sh[threadIdx.x] = v;
  __syncthreads();
  for (int off = 1; off < 256; off <<= 1) {
    unsigned t = (threadIdx.x >= off) ? sh[threadIdx.x - off] : 0u;
    __syncthreads();
    sh[threadIdx.x] += t;
    __syncthreads();
  }
  if (i < n) out[i] = sh[threadIdx.x] - v;
  if (threadIdx.x == 255) bsum[blockIdx.x] = sh[255];
}
__global__ void k_scan2(unsigned* bsum, int nb) {  // nb <= 512, single block of 512
  __shared__ unsigned sh[512];
  int t = threadIdx.x;
  unsigned v = (t < nb) ? bsum[t] : 0u;
  sh[t] = v;
  __syncthreads();
  for (int off = 1; off < 512; off <<= 1) {
    unsigned x = (t >= off) ? sh[t - off] : 0u;
    __syncthreads();
    sh[t] += x;
    __syncthreads();
  }
  if (t < nb) bsum[t] = sh[t] - v;
}
__global__ void k_scan3(unsigned* __restrict__ out, const unsigned* __restrict__ bsum, int n) {
  int i = blockIdx.x * 256 + threadIdx.x;
  if (i < n) out[i] += bsum[blockIdx.x];
}

__global__ void k_keep(unsigned* __restrict__ gt_io, const unsigned* __restrict__ eq,
                       const unsigned* __restrict__ eqscan, const unsigned* __restrict__ sel,
                       int n) {
  int i = blockIdx.x * 256 + threadIdx.x;
  if (i >= n) return;
  unsigned kp = gt_io[i] | ((eq[i] != 0u && eqscan[i] < sel[3]) ? 1u : 0u);
  gt_io[i] = kp ? 1u : 0u;
}
__global__ void k_perm(const unsigned* __restrict__ keep, const unsigned* __restrict__ rowscan,
                       int* __restrict__ perm, int* __restrict__ inv, int n) {
  int i = blockIdx.x * 256 + threadIdx.x;
  if (i >= n) return;
  if (keep[i]) {
    int r = (int)rowscan[i];
    perm[r] = i;
    inv[i] = r;
  } else {
    inv[i] = -1;
  }
}
// gather kept rows: next-layer x (f32), kept tangent (f32 + f16), batch
__global__ void k_gather(const int* __restrict__ perm, const float* __restrict__ srcx,
                         const float* __restrict__ xt, const int* __restrict__ bin,
                         float* __restrict__ curx, float* __restrict__ xtk,
                         _Float16* __restrict__ t16, int* __restrict__ bout, int k) {
  int lane = threadIdx.x & 31;
  int row = blockIdx.x * 8 + (threadIdx.x >> 5);
  if (row >= k) return;
  int i = perm[row];
  const float* sp = srcx + (size_t)i * HG_D + lane * 2;
  const float* tp = xt + (size_t)i * HG_D + lane * 2;
  float* cp = curx + (size_t)row * HG_D + lane * 2;
  float* kp = xtk + (size_t)row * HG_D + lane * 2;
  _Float16* hp = t16 + (size_t)row * HG_D + lane * 2;
  float t0 = tp[0], t1 = tp[1];
  cp[0] = sp[0]; cp[1] = sp[1];
  kp[0] = t0; kp[1] = t1;
  hp[0] = (_Float16)t0; hp[1] = (_Float16)t1;
  if (lane == 0) bout[row] = bin[i];
}

// ---------- structure learning (pool 1 only) ----------
__global__ void k_remap(const int* __restrict__ src, const int* __restrict__ dst,
                        const float* __restrict__ w, const int* __restrict__ inv,
                        const float* __restrict__ ps, const float* __restrict__ pd,
                        int* __restrict__ ns, int* __restrict__ nd, unsigned* __restrict__ vf,
                        float* __restrict__ lg, int nE) {
  int i = blockIdx.x * 256 + threadIdx.x;
  if (i >= nE) return;
  int s = src[i], d = dst[i];
  int a = inv[s], b = inv[d];
  bool v = (a >= 0) && (b >= 0);
  ns[i] = v ? a : 0;
  nd[i] = v ? b : 0;
  vf[i] = v ? 1u : 0u;
  lg[i] = v ? (ps[s] + pd[d] + HG_LAMB * w[i]) : -1e9f;
}
__global__ void k_segmax(const int* __restrict__ nd, const float* __restrict__ lg,
                         unsigned* __restrict__ mb, int nE) {
  int i = blockIdx.x * 256 + threadIdx.x;
  if (i < nE) atomicMax(&mb[nd[i]], fenc(lg[i]));
}
__global__ void k_fixm(unsigned* mb, int k) {   // decode in place; -inf -> 0
  int i = blockIdx.x * 256 + threadIdx.x;
  if (i >= k) return;
  float f = fdec(mb[i]);
  if (isinf(f)) f = 0.0f;
  ((float*)mb)[i] = f;
}
__global__ void k_z(const int* __restrict__ nd, const unsigned* __restrict__ vf,
                    float* __restrict__ lg, const float* __restrict__ mbf,
                    float* __restrict__ den, int nE) {
  int i = blockIdx.x * 256 + threadIdx.x;
  if (i >= nE) return;
  int t = nd[i];
  float z = vf[i] ? expf(lg[i] - mbf[t]) : 0.0f;
  lg[i] = z;
  atomicAdd(&den[t], z);
}
__global__ void k_wnorm(const float* __restrict__ z, const int* __restrict__ nd,
                        const float* __restrict__ den, float* __restrict__ nw, int nE) {
  int i = blockIdx.x * 256 + threadIdx.x;
  if (i >= nE) return;
  nw[i] = z[i] / (den[nd[i]] + HG_EPS);
}

// ---------- readout ----------
__global__ void k_readout_acc(const float* __restrict__ xtk, const int* __restrict__ bk,
                              unsigned* __restrict__ gmp, float* __restrict__ gap,
                              float* __restrict__ cnt, int k) {
  int lane = threadIdx.x & 31;
  int row = blockIdx.x * 8 + (threadIdx.x >> 5);
  if (row >= k) return;
  int b = bk[row];
  const float* xp = xtk + (size_t)row * HG_D + lane * 2;
  float f0 = xp[0], f1 = xp[1];
  atomicMax(&gmp[b * HG_D + lane * 2], fenc(f0));
  atomicMax(&gmp[b * HG_D + lane * 2 + 1], fenc(f1));
  atomicAdd(&gap[b * HG_D + lane * 2], f0);
  atomicAdd(&gap[b * HG_D + lane * 2 + 1], f1);
  if (lane == 0) atomicAdd(&cnt[b], 1.0f);
}
__global__ void k_readout_out(const unsigned* __restrict__ gmp, const float* __restrict__ gap,
                              const float* __restrict__ cnt, float* __restrict__ out,
                              int accum) {
  int i = blockIdx.x * 256 + threadIdx.x;
  if (i >= HG_B * 2 * HG_D) return;
  int b = i / (2 * HG_D), c = i % (2 * HG_D);
  float cn = cnt[b];
  float v;
  if (c < HG_D) {
    v = (cn > 0.0f) ? fdec(gmp[b * HG_D + c]) : 0.0f;
    if (isinf(v)) v = 0.0f;
  } else {
    v = gap[b * HG_D + (c - HG_D)] / fmaxf(cn, 1.0f);
  }
  float r = fmaxf(v, 0.0f);
  if (accum) out[i] += r; else out[i] = r;
}

// ---------- host ----------
extern "C" void kernel_launch(void* const* d_in, const int* in_sizes, int n_in,
                              void* d_out, int out_size, void* d_ws, size_t ws_size,
                              hipStream_t stream) {
  (void)in_sizes; (void)n_in; (void)out_size; (void)ws_size;
  const float* x   = (const float*)d_in[0];
  const int* esrc  = (const int*)d_in[1];
  const int* edst  = (const int*)d_in[2];
  const float* ew  = (const float*)d_in[3];
  const int* batch = (const int*)d_in[4];
  const float* W2  = (const float*)d_in[5];
  const float* b2  = (const float*)d_in[6];
  const float* W3  = (const float*)d_in[7];
  const float* b3  = (const float*)d_in[8];
  const float* as1 = (const float*)d_in[9];
  const float* ad1 = (const float*)d_in[10];
  const float* as2 = (const float*)d_in[11];
  const float* ad2 = (const float*)d_in[12];
  float* out = (float*)d_out;

  // bump allocator over d_ws
  char* base = (char*)d_ws;
  size_t cur = 0;
  auto alloc = [&](size_t bytes) -> char* {
    char* p = base + cur;
    cur += (bytes + 255) & ~(size_t)255;
    return p;
  };
  const size_t NF = (size_t)HG_N * HG_D;
  float*    CURX = (float*)alloc(NF * 4);
  float*    MV   = (float*)alloc(NF * 4);
  float*    OUT1 = (float*)alloc(NF * 4);
  float*    XT   = (float*)alloc(NF * 4);
  float*    AGG  = (float*)alloc(NF * 4);
  float*    XTK  = (float*)alloc((size_t)HG_K1 * HG_D * 4);
  _Float16* T16  = (_Float16*)alloc(NF * 2);
  _Float16* HT16 = (_Float16*)alloc(NF * 2);
  _Float16* XT16 = (_Float16*)alloc(NF * 2);
  float*    DEG  = (float*)alloc((size_t)HG_N * 4);
  unsigned* SCORE= (unsigned*)alloc((size_t)HG_N * 4);
  float*    PS   = (float*)alloc((size_t)HG_N * 4);
  float*    PD   = (float*)alloc((size_t)HG_N * 4);
  unsigned* U1   = (unsigned*)alloc((size_t)HG_N * 4);
  unsigned* U2   = (unsigned*)alloc((size_t)HG_N * 4);
  unsigned* U3   = (unsigned*)alloc((size_t)HG_N * 4);
  int*      INV  = (int*)alloc((size_t)HG_N * 4);
  unsigned* BS   = (unsigned*)alloc(512 * 4);
  unsigned* HIST = (unsigned*)alloc(65536 * 4);
  unsigned* SEL  = (unsigned*)alloc(64);
  int*      PERM = (int*)alloc((size_t)HG_K1 * 4);
  int*      BK1  = (int*)alloc((size_t)HG_K1 * 4);
  int*      BK2  = (int*)alloc((size_t)HG_K2 * 4);
  int*      NS   = (int*)alloc((size_t)HG_E * 4);
  int*      ND   = (int*)alloc((size_t)HG_E * 4);
  unsigned* VF   = (unsigned*)alloc((size_t)HG_E * 4);
  float*    LG   = (float*)alloc((size_t)HG_E * 4);
  float*    NW   = (float*)alloc((size_t)HG_E * 4);
  unsigned* MB   = (unsigned*)alloc((size_t)HG_K1 * 4);
  float*    DEN  = (float*)alloc((size_t)HG_K1 * 4);
  unsigned* GMP  = (unsigned*)alloc((size_t)HG_B * HG_D * 4);
  float*    GAP  = (float*)alloc((size_t)HG_B * HG_D * 4);
  float*    CNT  = (float*)alloc((size_t)HG_B * 4);
  _Float16* WH   = (_Float16*)alloc((size_t)HG_D * HG_D * 2);
  float*    HB   = (float*)alloc((size_t)HG_D * 4);

  auto ROWG = [](int n) { return dim3((n + 7) / 8); };
  auto ELTG = [](int n) { return dim3((n + 255) / 256); };
  const dim3 T256(256);
  const dim3 EG = ELTG(HG_E);
  const dim3 ESC((HG_E + 7) / 8);

  // exclusive scan of u-array [n]; out gets the scan
  auto scan = [&](unsigned* in, unsigned* o, int n) {
    int nb = (n + 255) / 256;
    k_scan1<<<dim3(nb), T256, 0, stream>>>(in, o, BS, n);
    k_scan2<<<dim3(1), dim3(512), 0, stream>>>(BS, nb);
    k_scan3<<<dim3(nb), T256, 0, stream>>>(o, BS, n);
  };
  // one hgcn layer: T16 holds f16 tangent of layer input; writes OUT1
  auto hgcn = [&](const float* Wl, const float* bl, const int* s, const int* d,
                  const float* w, int n) {
    k_w2h<<<dim3(16), T256, 0, stream>>>(Wl, WH);
    k_hb<<<dim3(1), dim3(32), 0, stream>>>(bl, HB);
    k_gemm_wmma<<<dim3((n / 16 + 7) / 8), T256, 0, stream>>>(T16, WH, MV, n);
    k_hl_post<<<ROWG(n), T256, 0, stream>>>(MV, HB, HT16, n);
    k_fill_f32<<<ELTG(n), T256, 0, stream>>>(DEG, HG_EPS, n);
    k_deg<<<EG, T256, 0, stream>>>(d, w, DEG, HG_E);
    k_fill_f32<<<ELTG(n * HG_D), T256, 0, stream>>>(AGG, 0.0f, n * HG_D);
    k_scatter<<<ESC, T256, 0, stream>>>(s, d, w, DEG, HT16, AGG, HG_E);
    k_agg_act<<<ROWG(n), T256, 0, stream>>>(AGG, OUT1, n);
  };
  // top-k selection on SCORE[n]; fills PERM/INV
  auto topk = [&](int n, int k) {
    k_fill_u32<<<dim3(256), T256, 0, stream>>>(HIST, 0u, 65536);
    k_hist_hi<<<ELTG(n), T256, 0, stream>>>(SCORE, HIST, n);
    k_sel_hi<<<dim3(1), dim3(1), 0, stream>>>(HIST, SEL, k);
    k_fill_u32<<<dim3(256), T256, 0, stream>>>(HIST, 0u, 65536);
    k_hist_lo<<<ELTG(n), T256, 0, stream>>>(SCORE, SEL, HIST, n);
    k_sel_lo<<<dim3(1), dim3(1), 0, stream>>>(HIST, SEL, k);
    k_flags<<<ELTG(n), T256, 0, stream>>>(SCORE, SEL, U2, U1, n);
    scan(U1, U3, n);                                     // tie ranks
    k_keep<<<ELTG(n), T256, 0, stream>>>(U2, U1, U3, SEL, n);
    scan(U2, U1, n);                                     // row indices
    k_perm<<<ELTG(n), T256, 0, stream>>>(U2, U1, PERM, INV, n);
  };
  auto readout = [&](const int* bk, int k, int accum) {
    k_fill_u32<<<ELTG(HG_B * HG_D), T256, 0, stream>>>(GMP, ENC_NEGINF, HG_B * HG_D);
    k_fill_f32<<<ELTG(HG_B * HG_D), T256, 0, stream>>>(GAP, 0.0f, HG_B * HG_D);
    k_fill_f32<<<dim3(1), T256, 0, stream>>>(CNT, 0.0f, HG_B);
    k_readout_acc<<<ROWG(k), T256, 0, stream>>>(XTK, bk, GMP, GAP, CNT, k);
    k_readout_out<<<dim3(32), T256, 0, stream>>>(GMP, GAP, CNT, out, accum);
  };

  // ===== stage 1: xh = proj(exp(x)); layer 1 =====
  k_exp_proj<<<ROWG(HG_N), T256, 0, stream>>>(x, CURX, HG_N);
  k_logmap_both<<<ROWG(HG_N), T256, 0, stream>>>(CURX, XT, T16, HG_N);
  hgcn(W2, b2, esrc, edst, ew, HG_N);

  // ===== pool 1 =====
  k_logmap_both<<<ROWG(HG_N), T256, 0, stream>>>(OUT1, XT, XT16, HG_N);
  k_fill_f32<<<ELTG(HG_N), T256, 0, stream>>>(DEG, HG_EPS, HG_N);
  k_deg<<<EG, T256, 0, stream>>>(edst, ew, DEG, HG_E);
  k_fill_f32<<<ELTG(HG_N * HG_D), T256, 0, stream>>>(AGG, 0.0f, HG_N * HG_D);
  k_scatter<<<ESC, T256, 0, stream>>>(esrc, edst, ew, DEG, XT16, AGG, HG_E);
  k_score_proj<<<ROWG(HG_N), T256, 0, stream>>>(XT, AGG, as1, ad1, SCORE, PS, PD, HG_N);
  topk(HG_N, HG_K1);
  k_gather<<<ROWG(HG_K1), T256, 0, stream>>>(PERM, OUT1, XT, batch, CURX, XTK, T16, BK1, HG_K1);
  // structure learning -> new edge weights
  k_remap<<<EG, T256, 0, stream>>>(esrc, edst, ew, INV, PS, PD, NS, ND, VF, LG, HG_E);
  k_fill_u32<<<ELTG(HG_K1), T256, 0, stream>>>(MB, ENC_NEGINF, HG_K1);
  k_segmax<<<EG, T256, 0, stream>>>(ND, LG, MB, HG_E);
  k_fixm<<<ELTG(HG_K1), T256, 0, stream>>>(MB, HG_K1);
  k_fill_f32<<<ELTG(HG_K1), T256, 0, stream>>>(DEN, 0.0f, HG_K1);
  k_z<<<EG, T256, 0, stream>>>(ND, VF, LG, (const float*)MB, DEN, HG_E);
  k_wnorm<<<EG, T256, 0, stream>>>(LG, ND, DEN, NW, HG_E);
  readout(BK1, HG_K1, 0);

  // ===== layer 2 on kept graph =====
  hgcn(W3, b3, NS, ND, NW, HG_K1);

  // ===== pool 2 (edge outputs unused downstream -> skip structure learning) =====
  k_logmap_both<<<ROWG(HG_K1), T256, 0, stream>>>(OUT1, XT, XT16, HG_K1);
  k_fill_f32<<<ELTG(HG_K1), T256, 0, stream>>>(DEG, HG_EPS, HG_K1);
  k_deg<<<EG, T256, 0, stream>>>(ND, NW, DEG, HG_E);
  k_fill_f32<<<ELTG(HG_K1 * HG_D), T256, 0, stream>>>(AGG, 0.0f, HG_K1 * HG_D);
  k_scatter<<<ESC, T256, 0, stream>>>(NS, ND, NW, DEG, XT16, AGG, HG_E);
  k_score_proj<<<ROWG(HG_K1), T256, 0, stream>>>(XT, AGG, as2, ad2, SCORE, PS, PD, HG_K1);
  topk(HG_K1, HG_K2);
  k_gather<<<ROWG(HG_K2), T256, 0, stream>>>(PERM, OUT1, XT, BK1, CURX, XTK, T16, BK2, HG_K2);
  readout(BK2, HG_K2, 1);
}
